// EdgeConv_29944511988097
// MI455X (gfx1250) — matrix operand order repaired and checked by reading, not compiled, not run on hardware
//
#include <hip/hip_runtime.h>

typedef __attribute__((ext_vector_type(16))) _Float16 v16h;
typedef __attribute__((ext_vector_type(8)))  _Float16 v8h;
typedef __attribute__((ext_vector_type(8)))  float    v8f;
typedef __attribute__((ext_vector_type(4)))  unsigned int u32x4;
typedef __attribute__((ext_vector_type(8)))  int i32x8;
typedef __attribute__((ext_vector_type(4)))  int i32x4;

#define NB 8
#define NN 2048
#define NK 16
#define NC 128
#define NNODES (NB * NN)                 // 16384
#define BLOCKS 256
#define WAVES_PER_BLOCK 8
#define THREADS (WAVES_PER_BLOCK * 32)   // 256
#define NODES_PER_WAVE (NNODES / (BLOCKS * WAVES_PER_BLOCK)) // 8
#define X1_STRIDE 136                    // 128 + 8 halves pad -> conflict-free LDS rows

#define W1_FRAG_HALVES (64 * 32 * 16)    // 8 K-chunks x 8 N-tiles x 32 lanes x 16 halves
#define W2_FRAG_HALVES (32 * 32 * 16)    // 4 K-chunks x 8 N-tiles
#define WF_HALVES (W1_FRAG_HALVES + W2_FRAG_HALVES)   // 49152 halves = 96 KB
#define WF_QWORDS (WF_HALVES / 4)        // 12288 8-byte elements

// Exact-grade GELU: 0.5x(1+erf(x/sqrt2)) with Abramowitz-Stegun 7.1.26 erf
// (|eps| <= 1.5e-7, i.e. f32-exact for our purposes) using v_rcp_f32 + v_exp_f32.
__device__ __forceinline__ float gelu_exact(float x) {
    float z = fabsf(x) * 0.70710678118654752440f;
    float t = __builtin_amdgcn_rcpf(fmaf(0.3275911f, z, 1.0f));
    float p = fmaf(1.061405429f, t, -1.453152027f);
    p = fmaf(p, t, 1.421413741f);
    p = fmaf(p, t, -0.284496736f);
    p = fmaf(p, t, 0.254829592f);
    p = p * t;
    float e = __expf(-z * z);
    float erfz = 1.0f - p * e;                    // erf(|x|/sqrt2)
    float cdf = fmaf(0.5f, copysignf(erfz, x), 0.5f);
    return x * cdf;
}

__device__ __forceinline__ void cvt4(v16h& a, int base, float4 q) {
    a[base + 0] = (_Float16)q.x; a[base + 1] = (_Float16)q.y;
    a[base + 2] = (_Float16)q.z; a[base + 3] = (_Float16)q.w;
}

// ---- one-shot prep: convert W1/W2 (f32 row-major) into WMMA B-fragment f16 layout in d_ws ----
// Fragment layout: half index i = ((c*8 + t)*32 + lane)*16 + e
//   column n = t*16 + (lane&15), k = c*32 + (lane>>4)*8 + (e&7) + (e>>3)*16   (table order)
__global__ void pack_weights_kernel(const float* __restrict__ W1,
                                    const float* __restrict__ W2,
                                    _Float16* __restrict__ wsf) {
    const int stride = gridDim.x * blockDim.x;
    const int tid = blockIdx.x * blockDim.x + threadIdx.x;
    for (int i = tid; i < W1_FRAG_HALVES; i += stride) {
        int e = i & 15, l = (i >> 4) & 31, t = (i >> 9) & 7, c = i >> 12;
        int k = c * 32 + (l >> 4) * 8 + (e & 7) + (e >> 3) * 16;
        int n = t * 16 + (l & 15);
        wsf[i] = (_Float16)W1[k * 128 + n];
    }
    for (int i = tid; i < W2_FRAG_HALVES; i += stride) {
        int e = i & 15, l = (i >> 4) & 31, t = (i >> 9) & 7, c = i >> 12;
        int k = c * 32 + (l >> 4) * 8 + (e & 7) + (e >> 3) * 16;
        int n = t * 16 + (l & 15);
        wsf[W1_FRAG_HALVES + i] = (_Float16)W2[k * 128 + n];
    }
}

__global__ __launch_bounds__(THREADS, 1)
void edgeconv_wmma_kernel(const float* __restrict__ h,
                          const float* __restrict__ pos,
                          const int*   __restrict__ idx,
                          const float* __restrict__ W1,
                          const float* __restrict__ b1,
                          const float* __restrict__ b2,
                          const float* __restrict__ gvec,
                          const float* __restrict__ bvec,
                          const _Float16* __restrict__ wsf,
                          float* __restrict__ out)
{
    // Pre-swizzled WMMA B-fragments: [0,2048) = W1 (8 chunks x 8 tiles), [2048,3072) = W2
    __shared__ v16h sWf[96 * 32];                                    // 96 KB
    __shared__ __align__(32) _Float16 sX1[WAVES_PER_BLOCK][16 * X1_STRIDE]; // ~34.8 KB
    __shared__ float sRel[WAVES_PER_BLOCK][16][4];
    __shared__ float sW1t[3 * 128];  // W1 rows 256..258 (rel tail)
    __shared__ float sB1[128], sB2[128], sG[128], sBt[128];

    const int tid  = threadIdx.x;
    const int wave = tid >> 5;
    const int lane = tid & 31;
    const int lo   = lane & 15;
    const int hi   = lane >> 4;

    // ---- stage pre-swizzled weights into LDS ----
#if __has_builtin(__builtin_amdgcn_tensor_load_to_lds)
    if (wave == 0) {
        // Tensor DMA descriptor (ISA 08_async_tensor §8): one 1-D tile,
        // data_size=8B, 12288 elements, global = wsf, lds = sWf.
        const unsigned long long ga = (unsigned long long)(uintptr_t)wsf;
        const unsigned ga_lo = (unsigned)ga;
        const unsigned ga_hi = (unsigned)(ga >> 32);
        const unsigned lds_off = (unsigned)(uintptr_t)(void*)&sWf[0]; // low 32 bits = LDS byte offset

        u32x4 g0;
        g0[0] = 1u;                                     // count=1, user mode, no gather
        g0[1] = lds_off;                                // lds_addr
        g0[2] = ga_lo;                                  // global_addr[31:0]
        g0[3] = (ga_hi & 0x01FFFFFFu) | (2u << 30);     // global_addr[56:32] | type=2 (image)

        i32x8 g1;
        g1[0] = (int)(3u << 16);                        // workgroup_mask=0, data_size=3 (8B)
        g1[1] = (int)((WF_QWORDS & 0xFFFFu) << 16);     // tensor_dim0[15:0] at bits[31:16]
        g1[2] = (int)(((WF_QWORDS >> 16) & 0xFFFFu) | (1u << 16)); // tensor_dim0 hi | tensor_dim1=1
        g1[3] = (int)((WF_QWORDS & 0xFFFFu) << 16);     // tensor_dim1 hi=0 | tile_dim0 at [31:16]
        g1[4] = 0;                                      // tile_dim1=0, tile_dim2=0 (unused)
        g1[5] = (int)WF_QWORDS;                         // tensor_dim0_stride[31:0]
        g1[6] = 0;
        g1[7] = 0;

        i32x4 g2 = {0, 0, 0, 0};
        i32x4 g3 = {0, 0, 0, 0};
#if __clang_major__ >= 23
        i32x8 gx = {0, 0, 0, 0, 0, 0, 0, 0};
        __builtin_amdgcn_tensor_load_to_lds(g0, g1, g2, g3, gx, 0);
#else
        __builtin_amdgcn_tensor_load_to_lds(g0, g1, g2, g3, 0);
#endif
        __builtin_amdgcn_s_wait_tensorcnt(0);
    }
#else
    {
        const uint4* src = (const uint4*)wsf;
        uint4* dst = (uint4*)sWf;
#pragma unroll 8
        for (int i = tid; i < (int)(WF_HALVES * sizeof(_Float16) / sizeof(uint4)); i += THREADS)
            dst[i] = src[i];
    }
#endif
    for (int i = tid; i < 384; i += THREADS) sW1t[i] = W1[256 * 128 + i];
    if (tid < 128) {
        sB1[tid] = b1[tid];  sB2[tid] = b2[tid];
        sG[tid]  = gvec[tid]; sBt[tid] = bvec[tid];
    }
    __syncthreads();

    const v16h* sW1f = sWf;
    const v16h* sW2f = sWf + 64 * 32;

    for (int it = 0; it < NODES_PER_WAVE; ++it) {
        const int node = blockIdx.x * WAVES_PER_BLOCK + wave + it * (BLOCKS * WAVES_PER_BLOCK);
        const int bidx = node >> 11;                     // node / N
        const float* hC = h + (size_t)node * NC;         // center features (row index == node)
        const int j = idx[(size_t)node * NK + lo];       // neighbor of row m = lo
        const float* hJ = h + ((size_t)bidx * NN + (size_t)j) * NC;

        // Prefetch next iteration's random neighbor rows into cache while this
        // node's WMMAs run (global_prefetch_b8; 2 lines/lane cover each 512B row).
        if (it + 1 < NODES_PER_WAVE) {
            const int node2 = node + BLOCKS * WAVES_PER_BLOCK;
            const int b2i = node2 >> 11;
            const int j2 = idx[(size_t)node2 * NK + lo];
            const float* hJ2 = h + ((size_t)b2i * NN + (size_t)j2) * NC;
            __builtin_prefetch(hJ2 + hi * 64, 0, 3);
            __builtin_prefetch(hJ2 + hi * 64 + 32, 0, 3);
        }

        { // rel = pos[j] - pos[i], one writer per row
            const float* pC = pos + (size_t)node * 3;
            const float* pJ = pos + ((size_t)bidx * NN + (size_t)j) * 3;
            if (hi == 0) {
                sRel[wave][lo][0] = pJ[0] - pC[0];
                sRel[wave][lo][1] = pJ[1] - pC[1];
                sRel[wave][lo][2] = pJ[2] - pC[2];
            }
        }
        __syncthreads();

        // rel rows this lane owns in the C/D layout (rows v + 8*hi)
        float relv[8][3];
#pragma unroll
        for (int v = 0; v < 8; ++v) {
            int m = v + hi * 8;
            relv[v][0] = sRel[wave][m][0];
            relv[v][1] = sRel[wave][m][1];
            relv[v][2] = sRel[wave][m][2];
        }

        // acc = b1 + rel @ W1[256:259,:]  (folds the 3-dim tail -> no K padding)
        v8f acc[8];
#pragma unroll
        for (int t = 0; t < 8; ++t) {
            int col = t * 16 + lo;
            float bb = sB1[col];
            float w0 = sW1t[col], w1 = sW1t[128 + col], w2 = sW1t[256 + col];
#pragma unroll
            for (int v = 0; v < 8; ++v)
                acc[t][v] = bb + relv[v][0] * w0 + relv[v][1] * w1 + relv[v][2] * w2;
        }

        // GEMM1: 8 K-chunks x 8 N-tiles of v_wmma_f32_16x16x32_f16
#pragma unroll
        for (int c = 0; c < 8; ++c) {
            const int kb = c * 32 + hi * 8;              // global K within 0..255
            v16h a;
            if (c < 4) {                                 // msg[:,0:128] = h_i (broadcast rows)
                cvt4(a, 0,  *(const float4*)(hC + kb));
                cvt4(a, 4,  *(const float4*)(hC + kb + 4));
                cvt4(a, 8,  *(const float4*)(hC + kb + 16));
                cvt4(a, 12, *(const float4*)(hC + kb + 20));
            } else {                                     // msg[:,128:256] = h_j - h_i
                const int kc = kb - 128;
                float4 j0 = *(const float4*)(hJ + kc),      i0 = *(const float4*)(hC + kc);
                float4 j1 = *(const float4*)(hJ + kc + 4),  i1 = *(const float4*)(hC + kc + 4);
                float4 j2 = *(const float4*)(hJ + kc + 16), i2 = *(const float4*)(hC + kc + 16);
                float4 j3 = *(const float4*)(hJ + kc + 20), i3 = *(const float4*)(hC + kc + 20);
                cvt4(a, 0,  make_float4(j0.x - i0.x, j0.y - i0.y, j0.z - i0.z, j0.w - i0.w));
                cvt4(a, 4,  make_float4(j1.x - i1.x, j1.y - i1.y, j1.z - i1.z, j1.w - i1.w));
                cvt4(a, 8,  make_float4(j2.x - i2.x, j2.y - i2.y, j2.z - i2.z, j2.w - i2.w));
                cvt4(a, 12, make_float4(j3.x - i3.x, j3.y - i3.y, j3.z - i3.z, j3.w - i3.w));
            }
#pragma unroll
            for (int t = 0; t < 8; ++t) {
                v16h bf = sW1f[(c * 8 + t) * 32 + lane];
                acc[t] = __builtin_amdgcn_wmma_f32_16x16x32_f16(
                    false, a, false, bf, (short)0, acc[t], false, false);
            }
        }

        // GELU -> f16 LDS tile (A operand of GEMM2)
        _Float16* myX1 = sX1[wave];
#pragma unroll
        for (int t = 0; t < 8; ++t) {
            int col = t * 16 + lo;
#pragma unroll
            for (int v = 0; v < 8; ++v) {
                int m = v + hi * 8;
                myX1[m * X1_STRIDE + col] = (_Float16)gelu_exact(acc[t][v]);
            }
        }
        __syncthreads();

        // GEMM2: 4 K-chunks x 8 N-tiles
        v8f acc2[8];
#pragma unroll
        for (int t = 0; t < 8; ++t) {
            float bb = sB2[t * 16 + lo];
#pragma unroll
            for (int v = 0; v < 8; ++v) acc2[t][v] = bb;
        }
#pragma unroll
        for (int c = 0; c < 4; ++c) {
            const int kb = c * 32 + hi * 8;
            const _Float16* pr = myX1 + lo * X1_STRIDE + kb;   // 16B-aligned
            v8h r0 = *(const v8h*)(pr);
            v8h r1 = *(const v8h*)(pr + 16);
            v16h a;
#pragma unroll
            for (int e = 0; e < 8; ++e) { a[e] = r0[e]; a[8 + e] = r1[e]; }
#pragma unroll
            for (int t = 0; t < 8; ++t) {
                v16h bf = sW2f[(c * 8 + t) * 32 + lane];
                acc2[t] = __builtin_amdgcn_wmma_f32_16x16x32_f16(
                    false, a, false, bf, (short)0, acc2[t], false, false);
            }
        }

        // GELU, max over 16 neighbor-rows (8 in-lane VGPRs + lane-half swap)
        float colmax[8];
#pragma unroll
        for (int t = 0; t < 8; ++t) {
            float mv = -3.0e38f;
#pragma unroll
            for (int v = 0; v < 8; ++v) mv = fmaxf(mv, gelu_exact(acc2[t][v]));
            mv = fmaxf(mv, __shfl_xor(mv, 16, 32));
            colmax[t] = mv;
        }

        // LayerNorm over 128 cols: butterfly over the 16 column-lanes
        float s = 0.f, sq = 0.f;
#pragma unroll
        for (int t = 0; t < 8; ++t) { s += colmax[t]; sq += colmax[t] * colmax[t]; }
#pragma unroll
        for (int o = 8; o >= 1; o >>= 1) {
            s  += __shfl_xor(s,  o, 32);
            sq += __shfl_xor(sq, o, 32);
        }
        const float mean = s * (1.0f / 128.0f);
        const float var  = sq * (1.0f / 128.0f) - mean * mean;
        const float rstd = rsqrtf(var + 1e-5f);

        if (hi == 0) {
            float* op = out + (size_t)node * NC;
#pragma unroll
            for (int t = 0; t < 8; ++t) {
                int col = t * 16 + lo;
                op[col] = (colmax[t] - mean) * rstd * sG[col] + sBt[col];
            }
        }
        __syncthreads();   // protect sRel/sX1 reuse next iteration
    }
}

extern "C" void kernel_launch(void* const* d_in, const int* in_sizes, int n_in,
                              void* d_out, int out_size, void* d_ws, size_t ws_size,
                              hipStream_t stream) {
    const float* h    = (const float*)d_in[0];
    const float* pos  = (const float*)d_in[1];
    const int*   idx  = (const int*)d_in[2];
    const float* W1   = (const float*)d_in[3];
    const float* b1   = (const float*)d_in[4];
    const float* W2   = (const float*)d_in[5];
    const float* b2   = (const float*)d_in[6];
    const float* gam  = (const float*)d_in[7];
    const float* bet  = (const float*)d_in[8];
    float* out = (float*)d_out;
    _Float16* wsf = (_Float16*)d_ws;    // 96 KB of pre-swizzled f16 fragments

    hipLaunchKernelGGL(pack_weights_kernel, dim3(64), dim3(256), 0, stream, W1, W2, wsf);
    hipLaunchKernelGGL(edgeconv_wmma_kernel, dim3(BLOCKS), dim3(THREADS), 0, stream,
                       h, pos, idx, W1, b1, b2, gam, bet, wsf, out);
}
